// CrossImpactAnalyzer_66924180406943
// MI455X (gfx1250) — matrix operand order repaired and checked by reading, not compiled, not run on hardware
//
#include <hip/hip_runtime.h>
#include <hip/hip_bf16.h>

typedef _Float16 v16h __attribute__((ext_vector_type(16)));
typedef _Float16 v8h  __attribute__((ext_vector_type(8)));
typedef float    v8f  __attribute__((ext_vector_type(8)));

#define NN 131072
#define EE 131072
#define HHDIM 256

// ---------------------------------------------------------------- utilities
__global__ __launch_bounds__(256) void k_zero(float* __restrict__ p, int n) {
  int i = blockIdx.x * 256 + threadIdx.x;
  if (i < n) p[i] = 0.f;
}

// Pack row-major f32 [K, Nout] weight matrix into wave32 WMMA B-fragment
// layout: tile (kt, nt), lane L holds column n = nt*16 + L%16, halfs j=0..15
// cover K = kt*32 + (L/16)*8 + j (j<8), kt*32 + 16 + (L/16)*8 + (j-8) (j>=8). Each lane's 16 halfs are contiguous (32 B).
__global__ __launch_bounds__(256) void k_pack_b(const float* __restrict__ src,
                                                _Float16* __restrict__ dst,
                                                int K, int Nout) {
  int t = blockIdx.x * 256 + threadIdx.x;
  int KT = K >> 5, NT = Nout >> 4;
  if (t >= KT * NT * 32) return;
  int lane = t & 31, tile = t >> 5;
  int kt = tile % KT, nt = tile / KT;
  int n  = (nt << 4) + (lane & 15);
  int kb = (kt << 5) + ((lane >> 4) << 3);
  _Float16* d = dst + ((size_t)((nt * KT + kt) << 5) + lane) * 16;
#pragma unroll
  for (int j = 0; j < 16; ++j) d[j] = (_Float16)src[(size_t)(kb + j + ((j >> 3) << 3)) * Nout + n];
}

// x = node_features @ emb_w + emb_b (K=5); also write f16 shadow xh
__global__ __launch_bounds__(256) void k_embed(const float* __restrict__ nf,
                                               const float* __restrict__ w,
                                               const float* __restrict__ b,
                                               float* __restrict__ x,
                                               _Float16* __restrict__ xh) {
  int i = blockIdx.x * 256 + threadIdx.x;   // N*256 threads exactly
  int n = i >> 8, d = i & 255;
  float acc = b[d];
#pragma unroll
  for (int k = 0; k < 5; ++k) acc = fmaf(nf[n * 5 + k], w[k * 256 + d], acc);
  x[i] = acc;
  xh[i] = (_Float16)acc;
}

// refresh f16 shadow after residual update
__global__ __launch_bounds__(256) void k_xh_update(const float* __restrict__ x,
                                                   _Float16* __restrict__ xh) {
  int i = blockIdx.x * 256 + threadIdx.x;
  xh[i] = (_Float16)x[i];
}

// a1[n,h] = dot(x[n], att[h, 0:256]); a2[n,h] = dot(x[n], att[h, 256:512])
// one wave32 per node
__global__ __launch_bounds__(256) void k_att_node(const float* __restrict__ x,
                                                  const float* __restrict__ attl,
                                                  float* __restrict__ a1,
                                                  float* __restrict__ a2) {
  int wid  = (blockIdx.x * 256 + threadIdx.x) >> 5;
  int lane = threadIdx.x & 31;
  const float* xr = x + (size_t)wid * 256;
  float p1[4] = {0.f, 0.f, 0.f, 0.f}, p2[4] = {0.f, 0.f, 0.f, 0.f};
#pragma unroll
  for (int j = 0; j < 8; ++j) {
    int d = lane + (j << 5);
    float xv = xr[d];
#pragma unroll
    for (int h = 0; h < 4; ++h) {
      p1[h] = fmaf(xv, attl[h * 512 + d], p1[h]);
      p2[h] = fmaf(xv, attl[h * 512 + 256 + d], p2[h]);
    }
  }
#pragma unroll
  for (int off = 16; off >= 1; off >>= 1) {
#pragma unroll
    for (int h = 0; h < 4; ++h) {
      p1[h] += __shfl_xor(p1[h], off, 32);
      p2[h] += __shfl_xor(p2[h], off, 32);
    }
  }
  if (lane == 0) {
#pragma unroll
    for (int h = 0; h < 4; ++h) { a1[wid * 4 + h] = p1[h]; a2[wid * 4 + h] = p2[h]; }
  }
}

// Per edge: leaky_relu(a1[row]+a2[col]) -> head softmax -> t = exp(.) in (1,e);
// accumulate segment denominators with atomics. (Segment max is unnecessary
// because head-softmax output lies in (0,1).)
__global__ __launch_bounds__(256) void k_edge_scores(const float* __restrict__ a1,
                                                     const float* __restrict__ a2,
                                                     const int* __restrict__ row,
                                                     const int* __restrict__ col,
                                                     float* __restrict__ tbuf,
                                                     float* __restrict__ denom) {
  int e = blockIdx.x * 256 + threadIdx.x;
  int r = row[e], c = col[e];
  float s[4];
#pragma unroll
  for (int h = 0; h < 4; ++h) {
    float v = a1[r * 4 + h] + a2[c * 4 + h];
    s[h] = (v >= 0.f) ? v : 0.01f * v;
  }
  float m = fmaxf(fmaxf(s[0], s[1]), fmaxf(s[2], s[3]));
  float ex[4], sum = 0.f;
#pragma unroll
  for (int h = 0; h < 4; ++h) { ex[h] = __expf(s[h] - m); sum += ex[h]; }
  float inv = 1.f / sum;
#pragma unroll
  for (int h = 0; h < 4; ++h) {
    float t = __expf(ex[h] * inv);
    tbuf[e * 4 + h] = t;
    atomicAdd(&denom[r * 4 + h], t);
  }
}

__global__ __launch_bounds__(256) void k_edge_alpha(float* __restrict__ tbuf,
                                                    const float* __restrict__ denom,
                                                    const int* __restrict__ row) {
  int e = blockIdx.x * 256 + threadIdx.x;
  int r = row[e];
#pragma unroll
  for (int h = 0; h < 4; ++h) tbuf[e * 4 + h] = tbuf[e * 4 + h] / denom[r * 4 + h];
}

// ------------------------------------------------------- layer GEMM (WMMA)
// out[e,:] = (alpha-scaled gathered xh) @ lin_w + lin_b + x   (residual fused)
// A[e, h*256+kk] = alpha[e,h] * xh[col[e], kk]  generated in-register.
// One wave per 16x16 D tile; K = 1024 -> 32 x v_wmma_f32_16x16x32_f16.
__global__ __launch_bounds__(256) void k_gat_gemm(const _Float16* __restrict__ xh,
                                                  const _Float16* __restrict__ Bp,
                                                  const int* __restrict__ col,
                                                  const float* __restrict__ alpha,
                                                  const float* __restrict__ lin_b,
                                                  float* __restrict__ x) {
  const int NT = 16, KT = 32;
  int wid  = (blockIdx.x * 256 + threadIdx.x) >> 5;
  int lane = threadIdx.x & 31;
  int nt = wid & (NT - 1);
  int mt = wid >> 4;

  int erow = (mt << 4) + (lane & 15);                 // A-row owned by this lane
  const _Float16* xrow = xh + (size_t)col[erow] * 256;
  _Float16 al[4];
#pragma unroll
  for (int h = 0; h < 4; ++h) al[h] = (_Float16)alpha[erow * 4 + h];
  int base = (lane < 16) ? 0 : 8;                     // 16-bit A fragment layout
  const _Float16* bp = Bp + ((size_t)(nt * KT) * 32 + lane) * 16;

  v8f acc = {};
  for (int kt = 0; kt < KT; ++kt) {
    int k0 = ((kt << 5) + base) & 255;                // column inside xh row
    _Float16 ah = al[kt >> 3];                        // head block = kt*32/256
    v8h lo = *(const v8h*)(xrow + k0);
    v8h hi = *(const v8h*)(xrow + k0 + 16);
    lo = lo * ah;
    hi = hi * ah;
    v16h a;
#pragma unroll
    for (int j = 0; j < 8; ++j) { a[j] = lo[j]; a[j + 8] = hi[j]; }
    v16h b = *(const v16h*)bp;
    bp += 512;                                        // next kt fragment
    __builtin_prefetch(bp, 0, 1);
    acc = __builtin_amdgcn_wmma_f32_16x16x32_f16(false, a, false, b,
                                                 (short)0, acc, false, false);
  }
  // D layout: lane L, vgpr r -> (M = r + 8*(L/16), N = L%16)
  int n0 = (nt << 4) + (lane & 15);
  int m0 = (mt << 4) + ((lane & 16) ? 8 : 0);
  float bd = lin_b[n0];
#pragma unroll
  for (int r = 0; r < 8; ++r) {
    size_t idx = (size_t)(m0 + r) * 256 + n0;
    x[idx] = acc[r] + bd + x[idx];                    // fused bias + residual
  }
}

// --------------------------------------------- edge predictor GEMM (WMMA)
// h1 = relu(concat(xh[row], xh[col]) @ ep_w1 + b1); partial dot with ep_w2
// reduced across the 16 N-lanes and atomically accumulated per edge.
__global__ __launch_bounds__(256) void k_pred_gemm(const _Float16* __restrict__ xh,
                                                   const _Float16* __restrict__ Bp,
                                                   const int* __restrict__ row,
                                                   const int* __restrict__ col,
                                                   const float* __restrict__ b1,
                                                   const float* __restrict__ w2,
                                                   float* __restrict__ ci_acc) {
  const int NT = 16, KT = 16;                         // K = 512
  int wid  = (blockIdx.x * 256 + threadIdx.x) >> 5;
  int lane = threadIdx.x & 31;
  int nt = wid & (NT - 1);
  int mt = wid >> 4;

  int erow = (mt << 4) + (lane & 15);
  const _Float16* xr = xh + (size_t)row[erow] * 256;
  const _Float16* xc = xh + (size_t)col[erow] * 256;
  int base = (lane < 16) ? 0 : 8;
  const _Float16* bp = Bp + ((size_t)(nt * KT) * 32 + lane) * 16;

  v8f acc = {};
  for (int kt = 0; kt < KT; ++kt) {
    const _Float16* src = (kt < 8) ? xr : xc;         // concat(x[row], x[col])
    int k0 = ((kt << 5) + base) & 255;
    v8h lo = *(const v8h*)(src + k0);
    v8h hi = *(const v8h*)(src + k0 + 16);
    v16h a;
#pragma unroll
    for (int j = 0; j < 8; ++j) { a[j] = lo[j]; a[j + 8] = hi[j]; }
    v16h b = *(const v16h*)bp;
    bp += 512;
    __builtin_prefetch(bp, 0, 1);
    acc = __builtin_amdgcn_wmma_f32_16x16x32_f16(false, a, false, b,
                                                 (short)0, acc, false, false);
  }
  int n0 = (nt << 4) + (lane & 15);
  int m0 = (mt << 4) + ((lane & 16) ? 8 : 0);
  float bb = b1[n0], ww = w2[n0];
  float p[8];
#pragma unroll
  for (int r = 0; r < 8; ++r) {
    float v = acc[r] + bb;
    v = (v > 0.f) ? v : 0.f;                          // relu
    p[r] = v * ww;                                    // partial dot with ep_w2
  }
#pragma unroll
  for (int off = 1; off < 16; off <<= 1) {            // reduce over 16 N-lanes
#pragma unroll
    for (int r = 0; r < 8; ++r) p[r] += __shfl_xor(p[r], off, 32);
  }
  if ((lane & 15) == 0) {
#pragma unroll
    for (int r = 0; r < 8; ++r) atomicAdd(&ci_acc[m0 + r], p[r]);
  }
}

__global__ __launch_bounds__(256) void k_sigmoid(const float* __restrict__ ci_acc,
                                                 const float* __restrict__ b2,
                                                 float* __restrict__ ci) {
  int e = blockIdx.x * 256 + threadIdx.x;
  float v = ci_acc[e] + b2[0];
  ci[e] = 1.f / (1.f + __expf(-v));
}

// ------------------------------------------------------------------- launch
extern "C" void kernel_launch(void* const* d_in, const int* in_sizes, int n_in,
                              void* d_out, int out_size, void* d_ws, size_t ws_size,
                              hipStream_t stream) {
  (void)in_sizes; (void)n_in; (void)out_size; (void)ws_size;
  const int N = NN, E = EE, H = HHDIM, L = 3;

  const float* nf     = (const float*)d_in[0];
  const int*   ei     = (const int*)d_in[1];
  const int*   row    = ei;
  const int*   col    = ei + E;
  const float* emb_w  = (const float*)d_in[2];
  const float* emb_b  = (const float*)d_in[3];
  const float* att    = (const float*)d_in[4];   // [L,4,512]
  const float* lin_w  = (const float*)d_in[5];   // [L,1024,256]
  const float* lin_b  = (const float*)d_in[6];   // [L,256]
  const float* ep_w1  = (const float*)d_in[7];   // [512,256]
  const float* ep_b1  = (const float*)d_in[8];
  const float* ep_w2  = (const float*)d_in[9];   // [256]
  const float* ep_b2  = (const float*)d_in[10];

  float* x  = (float*)d_out;          // [N,256] working + final output
  float* ci = x + (size_t)N * H;      // [E]

  char* ws = (char*)d_ws;
  size_t off = 0;
  auto walloc = [&](size_t bytes) -> char* {
    char* p = ws + off;
    off = (off + bytes + 255) & ~(size_t)255;
    return p;
  };
  _Float16* xh     = (_Float16*)walloc((size_t)N * H * 2);        // 64 MiB
  _Float16* BpL    = (_Float16*)walloc((size_t)L * 1024 * 256 * 2);
  _Float16* BpF    = (_Float16*)walloc((size_t)512 * 256 * 2);
  float*    a1     = (float*)walloc((size_t)N * 4 * 4);
  float*    a2     = (float*)walloc((size_t)N * 4 * 4);
  float*    tbuf   = (float*)walloc((size_t)E * 4 * 4);           // t -> alpha
  float*    denom  = (float*)walloc((size_t)L * N * 4 * 4);
  float*    ci_acc = (float*)walloc((size_t)E * 4);

  // pack weights into WMMA B-fragment layout (tiny, once per launch)
  for (int l = 0; l < L; ++l)
    k_pack_b<<<((1024 / 32) * (256 / 16) * 32) / 256, 256, 0, stream>>>(
        lin_w + (size_t)l * 1024 * 256, BpL + (size_t)l * 1024 * 256, 1024, 256);
  k_pack_b<<<((512 / 32) * (256 / 16) * 32) / 256, 256, 0, stream>>>(ep_w1, BpF, 512, 256);

  k_zero<<<(L * N * 4) / 256, 256, 0, stream>>>(denom, L * N * 4);
  k_zero<<<E / 256, 256, 0, stream>>>(ci_acc, E);

  k_embed<<<(N * H) / 256, 256, 0, stream>>>(nf, emb_w, emb_b, x, xh);

  const int gemm_waves = (N / 16) * (256 / 16);   // 131072 tiles, 1 wave each
  for (int l = 0; l < L; ++l) {
    k_att_node<<<N / 8, 256, 0, stream>>>(x, att + (size_t)l * 4 * 512, a1, a2);
    k_edge_scores<<<E / 256, 256, 0, stream>>>(a1, a2, row, col, tbuf,
                                               denom + (size_t)l * N * 4);
    k_edge_alpha<<<E / 256, 256, 0, stream>>>(tbuf, denom + (size_t)l * N * 4, row);
    k_gat_gemm<<<gemm_waves / 8, 256, 0, stream>>>(xh, BpL + (size_t)l * 1024 * 256,
                                                   col, tbuf,
                                                   lin_b + (size_t)l * 256, x);
    k_xh_update<<<(N * H) / 256, 256, 0, stream>>>(x, xh);
  }

  k_pred_gemm<<<gemm_waves / 8, 256, 0, stream>>>(xh, BpF, row, col, ep_b1, ep_w2, ci_acc);
  k_sigmoid<<<E / 256, 256, 0, stream>>>(ci_acc, ep_b2, ci);
}
